// SSMPhonotacticsModel_88270167867770
// MI455X (gfx1250) — compile-verified, matrix operand order checked by reading, not run on hardware
//
#include <hip/hip_runtime.h>
#include <hip/hip_bf16.h>

#define K_DIM 512
#define S_DIM 64
#define BSZ   512
#define T_LEN 64
#define TILE_B 16
#define THREADS 256

typedef __attribute__((ext_vector_type(16))) __bf16 v16bf;
typedef __attribute__((ext_vector_type(8)))  float  v8f;

union Frag {
    uint4 u[2];
    v16bf v;
};

// A-fragment (16x32 bf16, M=batch rows) from LDS, row-major [m][k], row stride K_DIM.
// Lane l (l<16): M=l, K=k0..k0+15 ; lane l>=16: M=l-16, K=k0+16..k0+31.
__device__ __forceinline__ v16bf load_a_frag_lds(const __bf16* xbf, int k0) {
    const int l  = threadIdx.x & 31;
    const int m  = l & 15;
    const int kb = k0 + ((l >> 4) << 4);
    const uint4* p = (const uint4*)(xbf + m * K_DIM + kb);
    Frag f;
    f.u[0] = p[0];
    f.u[1] = p[1];
    return f.v;
}

// B-fragment (32x16 bf16, K x N) from a global row-major [n][k] matrix (row stride K_DIM).
// Lane l (l<16): N=n0+l, K=k0..k0+15 ; lane l>=16: N=n0+l-16, K=k0+16..k0+31.
__device__ __forceinline__ v16bf load_b_frag_g(const __bf16* __restrict__ mat, int n0, int k0) {
    const int l  = threadIdx.x & 31;
    const int n  = n0 + (l & 15);
    const int kb = k0 + ((l >> 4) << 4);
    const uint4* p = (const uint4*)(mat + n * K_DIM + kb);
    Frag f;
    f.u[0] = p[0];
    f.u[1] = p[1];
    return f.v;
}

// One-time preprocessing: sigmoid(pi), C*pi -> bf16, A -> bf16, B^T.
__global__ void ssm_prep_kernel(const float* __restrict__ A,
                                const float* __restrict__ B,
                                const float* __restrict__ C,
                                const float* __restrict__ piP,
                                __bf16* __restrict__ A_bf,
                                __bf16* __restrict__ Ceff_bf,
                                float* __restrict__ pi_g,
                                float* __restrict__ Btr_g) {
    const int idx = blockIdx.x * blockDim.x + threadIdx.x;
    if (idx < K_DIM * K_DIM) {
        A_bf[idx] = (__bf16)A[idx];
    }
    if (idx < S_DIM * K_DIM) {
        const int s = idx / K_DIM;
        const int k = idx - s * K_DIM;
        const float p = 1.0f / (1.0f + __expf(-piP[idx]));
        pi_g[idx]    = p;
        Ceff_bf[idx] = (__bf16)(C[idx] * p);
        Btr_g[idx]   = B[k * S_DIM + s];   // Bt[s][k] = B[k][s]
    }
}

__global__ __launch_bounds__(THREADS) void ssm_main_kernel(
        const __bf16* __restrict__ A_bf,    // K x K row-major [n][k]
        const __bf16* __restrict__ Ceff_bf, // S x K row-major [s][k]
        const float*  __restrict__ pi_g,    // S x K
        const float*  __restrict__ Btr_g,   // S x K
        const float*  __restrict__ init,    // K
        const int*    __restrict__ tokens,  // BSZ x T
        float*        __restrict__ out)     // BSZ
{
    __shared__ __bf16 xbf[TILE_B * K_DIM];      // 16 KB  bf16 shadow of state
    __shared__ float  xf[TILE_B * K_DIM];       // 32 KB  fp32 state
    __shared__ float  logits_sh[TILE_B * S_DIM];// 4 KB
    __shared__ float  ll_sh[TILE_B];
    __shared__ int    tok_sh[TILE_B];

    const int tid  = threadIdx.x;
    const int wave = tid >> 5;
    const int lane = tid & 31;
    const int b0   = blockIdx.x * TILE_B;

    // --- init state x = broadcast(init) ---
    for (int k = tid; k < K_DIM; k += THREADS) {
        const float v = init[k];
        const __bf16 vb = (__bf16)v;
        #pragma unroll
        for (int m = 0; m < TILE_B; ++m) {
            xf[m * K_DIM + k]  = v;
            xbf[m * K_DIM + k] = vb;
        }
    }
    if (tid < TILE_B) ll_sh[tid] = 0.0f;
    __syncthreads();

    #pragma unroll 1
    for (int t = 0; t < T_LEN; ++t) {
        if (tid < TILE_B) tok_sh[tid] = tokens[(b0 + tid) * T_LEN + t];

        // ---- Phase 1: logits = x @ Ceff^T  (16 x 64), waves 0..3 each own one N-tile ----
        if (wave < 4) {
            const int n0 = wave * 16;
            v8f acc = {};
            #pragma unroll 2
            for (int ks = 0; ks < K_DIM; ks += 32) {
                v16bf a = load_a_frag_lds(xbf, ks);
                v16bf b = load_b_frag_g(Ceff_bf, n0, ks);
                acc = __builtin_amdgcn_wmma_f32_16x16x32_bf16(
                        false, a, false, b, (short)0, acc, false, false);
            }
            // D layout: lane = n + 16*(m>=8), vgpr r -> m = r (+8 for upper half)
            const int n     = n0 + (lane & 15);
            const int mBase = (lane >> 4) << 3;
            #pragma unroll
            for (int r = 0; r < 8; ++r)
                logits_sh[(mBase + r) * S_DIM + n] = acc[r];
        }
        __syncthreads();

        // ---- Phase 2: log-softmax + gather ll ; wave w owns rows 2w, 2w+1 ----
        #pragma unroll
        for (int rr = 0; rr < 2; ++rr) {
            const int row = wave * 2 + rr;
            float v0 = logits_sh[row * S_DIM + lane];
            float v1 = logits_sh[row * S_DIM + lane + 32];
            float mx = fmaxf(v0, v1);
            #pragma unroll
            for (int off = 16; off > 0; off >>= 1)
                mx = fmaxf(mx, __shfl_xor(mx, off, 32));
            float se = __expf(v0 - mx) + __expf(v1 - mx);
            #pragma unroll
            for (int off = 16; off > 0; off >>= 1)
                se += __shfl_xor(se, off, 32);
            if (lane == 0) {
                const int tk = tok_sh[row];
                ll_sh[row] += logits_sh[row * S_DIM + tk] - mx - __logf(se);
            }
        }

        // ---- Phase 3: update = x @ A^T ; each wave owns 4 N-tiles (64 cols) ----
        v8f acc[4] = {};
        #pragma unroll 2
        for (int ks = 0; ks < K_DIM; ks += 32) {
            v16bf a = load_a_frag_lds(xbf, ks);
            #pragma unroll
            for (int j = 0; j < 4; ++j) {
                const int n0 = (wave * 4 + j) * 16;
                v16bf b = load_b_frag_g(A_bf, n0, ks);
                acc[j] = __builtin_amdgcn_wmma_f32_16x16x32_bf16(
                            false, a, false, b, (short)0, acc[j], false, false);
            }
        }
        __syncthreads();   // all reads of x (phase 1 + phase 3) complete

        // ---- Phase 4: gated blend x_new = (1-p)*x + p*(update + Bt[tok]) ----
        {
            const int nIn   = lane & 15;
            const int mBase = (lane >> 4) << 3;
            #pragma unroll
            for (int j = 0; j < 4; ++j) {
                const int n = (wave * 4 + j) * 16 + nIn;
                #pragma unroll
                for (int r = 0; r < 8; ++r) {
                    const int m  = mBase + r;
                    const int tk = tok_sh[m];
                    const float p  = pi_g[tk * K_DIM + n];
                    const float bt = Btr_g[tk * K_DIM + n];
                    const float xo = xf[m * K_DIM + n];
                    const float xn = (1.0f - p) * xo + p * (acc[j][r] + bt);
                    xf[m * K_DIM + n]  = xn;
                    xbf[m * K_DIM + n] = (__bf16)xn;
                }
            }
        }
        __syncthreads();   // new x visible before next step's reads
    }

    if (tid < TILE_B) out[b0 + tid] = ll_sh[tid];
}

extern "C" void kernel_launch(void* const* d_in, const int* in_sizes, int n_in,
                              void* d_out, int out_size, void* d_ws, size_t ws_size,
                              hipStream_t stream) {
    (void)in_sizes; (void)n_in; (void)out_size; (void)ws_size;

    const float* A      = (const float*)d_in[0];  // (K, K)
    const float* B      = (const float*)d_in[1];  // (K, S)
    const float* C      = (const float*)d_in[2];  // (S, K)
    const float* piP    = (const float*)d_in[3];  // (S, K)
    const float* init   = (const float*)d_in[4];  // (K,)
    const int*   tokens = (const int*)d_in[5];    // (BSZ, T)

    char* ws = (char*)d_ws;
    size_t off = 0;
    __bf16* A_bf   = (__bf16*)(ws + off); off += (size_t)K_DIM * K_DIM * sizeof(__bf16); // 512 KB
    __bf16* Ceff   = (__bf16*)(ws + off); off += (size_t)S_DIM * K_DIM * sizeof(__bf16); //  64 KB
    float*  pi_g   = (float*) (ws + off); off += (size_t)S_DIM * K_DIM * sizeof(float);  // 128 KB
    float*  Btr_g  = (float*) (ws + off); off += (size_t)S_DIM * K_DIM * sizeof(float);  // 128 KB

    const int prepThreads = 256;
    const int prepBlocks  = (K_DIM * K_DIM + prepThreads - 1) / prepThreads;
    hipLaunchKernelGGL(ssm_prep_kernel, dim3(prepBlocks), dim3(prepThreads), 0, stream,
                       A, B, C, piP, A_bf, Ceff, pi_g, Btr_g);

    hipLaunchKernelGGL(ssm_main_kernel, dim3(BSZ / TILE_B), dim3(THREADS), 0, stream,
                       A_bf, Ceff, pi_g, Btr_g, init, tokens, (float*)d_out);
}
